// Seq2Seq_24644522344757
// MI455X (gfx1250) — matrix operand order repaired and checked by reading, not compiled, not run on hardware
//
#include <hip/hip_runtime.h>
#include <hip/hip_bf16.h>

// ---------------------------------------------------------------------------
// Seq2Seq (LSTM encoder + Bahdanau attention decoder) for MI455X / gfx1250.
// Dense math: v_wmma_f32_16x16x32_bf16, operands fed from LDS with
// ds_load_b128 for both A and B fragments.
// Workspace requirement: ~147 MB.
// ---------------------------------------------------------------------------

typedef unsigned short u16;
typedef __attribute__((ext_vector_type(16))) __bf16   v16bf;
typedef __attribute__((ext_vector_type(8)))  float    v8f;
typedef __attribute__((ext_vector_type(4)))  unsigned u32x4;

#define V_  32001
#define H_  1024
#define S_  128
#define B_  64
#define A_  512
#define ML_ 32

__device__ __forceinline__ u16 f2bf(float f) {
    unsigned u = __builtin_bit_cast(unsigned, f);
    unsigned r = (u + 0x7FFFu + ((u >> 16) & 1u)) >> 16;
    return (u16)r;
}
__device__ __forceinline__ float bf2f(u16 s) {
    unsigned u = ((unsigned)s) << 16;
    return __builtin_bit_cast(float, u);
}
__device__ __forceinline__ float sigf(float x) { return 1.0f / (1.0f + expf(-x)); }

// Low 32 bits of a flat pointer into __shared__ == LDS byte offset
// (ISA 10.2: LDS aperture lives in addr[63:32]).
__device__ __forceinline__ unsigned ldsa(const void* p) {
    return (unsigned)(unsigned long long)p;
}

// ---------------------------------------------------------------------------
// Generic GEMM: C[M,N] = A_bf16[M,K](lda) @ W_bf16[N,K](ldw)^T (+bias)(+Csrc)
// Block tile 64x64, K-chunk 64, 256 threads = 8 waves, wave -> 16x32 outputs,
// 4 v_wmma_f32_16x16x32_bf16 per wave per K-chunk.
// ---------------------------------------------------------------------------
#define BM 64
#define BN 64
#define BK 64

union Frag16 { u32x4 q[2]; v16bf v; };

__global__ __launch_bounds__(256) void gemm_bf16(
    const u16* __restrict__ A, int lda,
    const u16* __restrict__ W, int ldw,
    const float* __restrict__ bias,
    const float* __restrict__ Csrc, int ldsrc,
    float* __restrict__ C, int ldc,
    int M, int N, int K)
{
    __shared__ __align__(16) u16 As[BM * BK];   // row-major M x K tile
    __shared__ __align__(16) u16 Ws[BN * BK];   // W rows = B columns (col-major B)

    const int tid  = threadIdx.x;
    const int lane = tid & 31;
    const int wid  = tid >> 5;
    const int m0   = blockIdx.y * BM;
    const int n0   = blockIdx.x * BN;
    const int wm   = (wid & 3) * 16;   // wave M offset inside tile
    const int wn   = (wid >> 2) * 32;  // wave N offset inside tile

    const int lrow = tid >> 2;         // 0..63
    const int lcol = (tid & 3) * 16;   // 0,16,32,48 (16 bf16 = 2 x 16B per thread)

    // Per-lane LDS fragment addresses (constant across the K loop).
    // A: lane l -> row M=wm+(l%16); chunks K=[kb,kb+8) and K=[16+kb,16+kb+8).
    const unsigned aA  = ldsa(&As[(wm + (lane & 15)) * BK]) + (lane >> 4) * 16;
    // B: ds_load_b128, lane l -> column N=wn+(l%16); chunks K=[kb,kb+8) and K=[16+kb,16+kb+8).
    const unsigned aB0 = ldsa(&Ws[(wn +      (lane & 15)) * BK]) + (lane >> 4) * 16;
    const unsigned aB1 = ldsa(&Ws[(wn + 16 + (lane & 15)) * BK]) + (lane >> 4) * 16;

    v8f acc0 = {};
    v8f acc1 = {};

    for (int k0 = 0; k0 < K; k0 += BK) {
        // ---- stage tiles (128-bit global loads -> 128-bit LDS stores) ----
        const u16* ga = A + (long)(m0 + lrow) * lda + k0 + lcol;
        *(u32x4*)&As[lrow * BK + lcol]     = *(const u32x4*)ga;
        *(u32x4*)&As[lrow * BK + lcol + 8] = *(const u32x4*)(ga + 8);

        const int nrow = n0 + lrow;
        u32x4 w0 = {}, w1 = {};
        const u16* gw = W + (long)nrow * ldw + k0 + lcol;
        if (nrow < N) { w0 = *(const u32x4*)gw; w1 = *(const u32x4*)(gw + 8); }
        *(u32x4*)&Ws[lrow * BK + lcol]     = w0;
        *(u32x4*)&Ws[lrow * BK + lcol + 8] = w1;

        if (k0 + BK < K) {                      // near-cache prefetch of next K tile
            __builtin_prefetch(ga + BK, 0, 3);
            if (nrow < N) __builtin_prefetch(gw + BK, 0, 3);
        }
        __syncthreads();

        // ---- fragment loads: ds_load_b128 (A and B) ----
        // Single asm block so the s_wait_dscnt is ordered before the WMMAs
        // (these DS ops are invisible to the compiler's counter tracking).
        Frag16 a0, a1, b00, b01, b10, b11;      // *_k0 / *_k1 halves of BK=64
        asm volatile(
            "ds_load_b128      %0,  %12\n\t"            // A  k[ 0:31] lo
            "ds_load_b128      %1,  %12 offset:32\n\t"  // A  k[ 0:31] hi
            "ds_load_b128      %2,  %12 offset:64\n\t"  // A  k[32:63] lo
            "ds_load_b128      %3,  %12 offset:96\n\t"  // A  k[32:63] hi
            "ds_load_b128      %4,  %13\n\t"            // B0 k[ 0:31] lo
            "ds_load_b128      %5,  %13 offset:32\n\t"  // B0 k[ 0:31] hi
            "ds_load_b128      %6,  %13 offset:64\n\t"  // B0 k[32:63] lo
            "ds_load_b128      %7,  %13 offset:96\n\t"  // B0 k[32:63] hi
            "ds_load_b128      %8,  %14\n\t"            // B1 k[ 0:31] lo
            "ds_load_b128      %9,  %14 offset:32\n\t"  // B1 k[ 0:31] hi
            "ds_load_b128      %10, %14 offset:64\n\t"  // B1 k[32:63] lo
            "ds_load_b128      %11, %14 offset:96\n\t"  // B1 k[32:63] hi
            "s_wait_dscnt 0x0"
            : "=v"(a0.q[0]), "=v"(a0.q[1]), "=v"(a1.q[0]), "=v"(a1.q[1]),
              "=v"(b00.q[0]), "=v"(b00.q[1]), "=v"(b01.q[0]), "=v"(b01.q[1]),
              "=v"(b10.q[0]), "=v"(b10.q[1]), "=v"(b11.q[0]), "=v"(b11.q[1])
            : "v"(aA), "v"(aB0), "v"(aB1));

        acc0 = __builtin_amdgcn_wmma_f32_16x16x32_bf16(false, a0.v, false, b00.v,
                                                       (short)0, acc0, false, false);
        acc1 = __builtin_amdgcn_wmma_f32_16x16x32_bf16(false, a0.v, false, b10.v,
                                                       (short)0, acc1, false, false);
        acc0 = __builtin_amdgcn_wmma_f32_16x16x32_bf16(false, a1.v, false, b01.v,
                                                       (short)0, acc0, false, false);
        acc1 = __builtin_amdgcn_wmma_f32_16x16x32_bf16(false, a1.v, false, b11.v,
                                                       (short)0, acc1, false, false);
        __syncthreads();
    }

    // C/D layout: VGPR r -> M = r + 8*(lane>=16), N = lane%16
    const int mloc = wm + (lane >> 4) * 8;
    const int nl   = lane & 15;
#pragma unroll
    for (int r = 0; r < 8; ++r) {
        const int gm  = m0 + mloc + r;
        const int gn0 = n0 + wn + nl;
        const int gn1 = gn0 + 16;
        if (gn0 < N) {
            float v = acc0[r];
            if (bias) v += bias[gn0];
            if (Csrc) v += Csrc[(long)gm * ldsrc + gn0];
            C[(long)gm * ldc + gn0] = v;
        }
        if (gn1 < N) {
            float v = acc1[r];
            if (bias) v += bias[gn1];
            if (Csrc) v += Csrc[(long)gm * ldsrc + gn1];
            C[(long)gm * ldc + gn1] = v;
        }
    }
}

// ---------------------------------------------------------------------------
// Small elementwise / prep kernels
// ---------------------------------------------------------------------------
__global__ void cvt_bf16(const float* __restrict__ s, u16* __restrict__ d, long n) {
    long i  = (long)blockIdx.x * 256 + threadIdx.x;
    long st = (long)gridDim.x * 256;
    for (; i < n; i += st) d[i] = f2bf(s[i]);
}

// Wcomb_e[n, 0:H] = Wih_e[n], Wcomb_e[n, H:2H] = Whh_e[n]   (n < 4H)
__global__ void build_wcomb(const float* __restrict__ wih, const float* __restrict__ whh,
                            u16* __restrict__ d) {
    long i  = (long)blockIdx.x * 256 + threadIdx.x;
    long st = (long)gridDim.x * 256;
    const long n = 4L * H_ * 2 * H_;
    for (; i < n; i += st) {
        long row = i >> 11;          // /2048
        long k   = i & 2047;
        float v  = (k < H_) ? wih[row * H_ + k] : whh[row * H_ + (k - H_)];
        d[i] = f2bf(v);
    }
}

__global__ void addv(const float* a, const float* b, float* o, int n) {
    int i = blockIdx.x * 256 + threadIdx.x;
    if (i < n) o[i] = a[i] + b[i];
}

__global__ void init_enc(float* __restrict__ c, u16* __restrict__ inpe) {
    int b = blockIdx.x, tid = threadIdx.x;
#pragma unroll
    for (int r = 0; r < 4; ++r) {
        int j = tid + r * 256;
        c[(long)b * H_ + j] = 0.0f;
        inpe[(long)b * 2 * H_ + H_ + j] = 0;  // h0 = 0 (bf16)
    }
}

__global__ void enc_prep(const int* __restrict__ tok_enc, const float* __restrict__ emb,
                         u16* __restrict__ inpe, int t) {
    int b = blockIdx.x, tid = threadIdx.x;
    int tk = tok_enc[b * S_ + t];
    const float* row = emb + (long)tk * H_;
#pragma unroll
    for (int r = 0; r < 4; ++r) {
        int j = tid + r * 256;
        inpe[(long)b * 2 * H_ + j] = f2bf(row[j]);
    }
}

// gates layout [B, 4H] = [i | f | g | o]; update c in place, write h (bf16)
// into next-step GEMM operand and into encoder_out[b, t, :].
__global__ void lstm_cell_enc(const float* __restrict__ g, float* __restrict__ c,
                              u16* __restrict__ inpe, u16* __restrict__ encb, int t) {
    int b = blockIdx.x, tid = threadIdx.x;
    const float* gb = g + (long)b * 4 * H_;
#pragma unroll
    for (int r = 0; r < 4; ++r) {
        int j = tid + r * 256;
        float ig = sigf(gb[j]);
        float fg = sigf(gb[H_ + j]);
        float gg = tanhf(gb[2 * H_ + j]);
        float og = sigf(gb[3 * H_ + j]);
        float cn = fg * c[(long)b * H_ + j] + ig * gg;
        c[(long)b * H_ + j] = cn;
        u16 h = f2bf(og * tanhf(cn));
        inpe[(long)b * 2 * H_ + H_ + j]   = h;
        encb[((long)b * S_ + t) * H_ + j] = h;
    }
}

__global__ void dec_init(const int* __restrict__ itok, const float* __restrict__ ictx,
                         int* __restrict__ tok, u16* __restrict__ inpd) {
    int b = blockIdx.x, tid = threadIdx.x;
    if (tid == 0) tok[b] = itok[b];
#pragma unroll
    for (int r = 0; r < 4; ++r) {
        int j = tid + r * 256;
        inpd[(long)b * 2 * H_ + H_ + j] = f2bf(ictx[(long)b * H_ + j]);
    }
}

__global__ void dec_prep(const int* __restrict__ tok, const float* __restrict__ emb,
                         u16* __restrict__ inpd) {
    int b = blockIdx.x, tid = threadIdx.x;
    int tk = tok[b];
    const float* row = emb + (long)tk * H_;
#pragma unroll
    for (int r = 0; r < 4; ++r) {
        int j = tid + r * 256;
        inpd[(long)b * 2 * H_ + j] = f2bf(row[j]);
    }
}

// decoder cell: state is reset to (hN,cN) each step -> cN is read-only.
__global__ void dec_cell(const float* __restrict__ g, const float* __restrict__ cN,
                         u16* __restrict__ hd) {
    int b = blockIdx.x, tid = threadIdx.x;
    const float* gb = g + (long)b * 4 * H_;
#pragma unroll
    for (int r = 0; r < 4; ++r) {
        int j = tid + r * 256;
        float ig = sigf(gb[j]);
        float fg = sigf(gb[H_ + j]);
        float gg = tanhf(gb[2 * H_ + j]);
        float og = sigf(gb[3 * H_ + j]);
        float cn = fg * cN[(long)b * H_ + j] + ig * gg;
        hd[(long)b * H_ + j] = f2bf(og * tanhf(cn));
    }
}

// Bahdanau attention for one decoder step. One block per batch row.
// score[s] = Wv . tanh(s1[b,s,:] + s2[b,:]) + bv ; softmax over S; context.
__global__ __launch_bounds__(256) void attention_kernel(
    const float* __restrict__ s1, const float* __restrict__ s2,
    const float* __restrict__ Wv, const float* __restrict__ bv,
    const u16* __restrict__ encb, u16* __restrict__ inpd)
{
    int b = blockIdx.x, tid = threadIdx.x;
    int lane = tid & 31, wid = tid >> 5;
    __shared__ float sc[S_];
    __shared__ float red[S_];

    // scores: each wave owns 16 sequence positions, lane-parallel over A
    for (int q = 0; q < 16; ++q) {
        int s = wid * 16 + q;
        const float* s1p = s1 + ((long)b * S_ + s) * A_;
        const float* s2p = s2 + (long)b * A_;
        float sum = 0.0f;
        for (int a = lane; a < A_; a += 32)
            sum += Wv[a] * tanhf(s1p[a] + s2p[a]);
#pragma unroll
        for (int off = 16; off; off >>= 1) sum += __shfl_xor(sum, off, 32);
        if (lane == 0) sc[s] = sum + bv[0];
    }
    __syncthreads();

    // softmax over S_=128
    if (tid < S_) red[tid] = sc[tid];
    __syncthreads();
    for (int off = 64; off >= 1; off >>= 1) {
        if (tid < off) red[tid] = fmaxf(red[tid], red[tid + off]);
        __syncthreads();
    }
    float mx = red[0];
    __syncthreads();
    if (tid < S_) sc[tid] = expf(sc[tid] - mx);
    __syncthreads();
    if (tid < S_) red[tid] = sc[tid];
    __syncthreads();
    for (int off = 64; off >= 1; off >>= 1) {
        if (tid < off) red[tid] += red[tid + off];
        __syncthreads();
    }
    float inv = 1.0f / red[0];
    __syncthreads();
    if (tid < S_) sc[tid] *= inv;
    __syncthreads();

    // context = sum_s attn[s] * encoder_out[b,s,:]; write bf16 straight into
    // the next decoder-step GEMM operand (inpd[:, H:2H]).
#pragma unroll
    for (int r = 0; r < 4; ++r) {
        int col = tid + r * 256;
        const u16* ep = encb + (long)b * S_ * H_ + col;
        float acc = 0.0f;
        for (int s = 0; s < S_; ++s) acc += sc[s] * bf2f(ep[(long)s * H_]);
        inpd[(long)b * 2 * H_ + H_ + col] = f2bf(acc);
    }
}

// argmax over V (first-max tie rule, matching jnp.argmax)
__global__ __launch_bounds__(256) void argmax_kernel(
    const float* __restrict__ logits, long ldc, int* __restrict__ tok, int V)
{
    int b = blockIdx.x, tid = threadIdx.x;
    const float* p = logits + (long)b * ldc;
    float best = -__builtin_inff();
    int   bi   = V;
    for (int v = tid; v < V; v += 256) {
        float x = p[v];
        if (x > best) { best = x; bi = v; }
    }
    __shared__ float bv_[256];
    __shared__ int   bix[256];
    bv_[tid] = best; bix[tid] = bi;
    __syncthreads();
    for (int off = 128; off >= 1; off >>= 1) {
        if (tid < off) {
            float o = bv_[tid + off]; int oi = bix[tid + off];
            if (o > bv_[tid] || (o == bv_[tid] && oi < bix[tid])) { bv_[tid] = o; bix[tid] = oi; }
        }
        __syncthreads();
    }
    if (tid == 0) tok[b] = bix[0];
}

// ---------------------------------------------------------------------------
// Host orchestration
// ---------------------------------------------------------------------------
extern "C" void kernel_launch(void* const* d_in, const int* in_sizes, int n_in,
                              void* d_out, int out_size, void* d_ws, size_t ws_size,
                              hipStream_t stream) {
    (void)in_sizes; (void)n_in; (void)out_size; (void)ws_size;

    const int*   tok_enc  = (const int*)d_in[0];
    const float* init_ctx = (const float*)d_in[1];
    const int*   init_tok = (const int*)d_in[2];
    const float* emb_enc  = (const float*)d_in[3];
    const float* emb_dec  = (const float*)d_in[4];
    const float* Wih_e    = (const float*)d_in[5];
    const float* Whh_e    = (const float*)d_in[6];
    const float* bih_e    = (const float*)d_in[7];
    const float* bhh_e    = (const float*)d_in[8];
    const float* Wih_d    = (const float*)d_in[9];
    const float* Whh_d    = (const float*)d_in[10];
    const float* bih_d    = (const float*)d_in[11];
    const float* bhh_d    = (const float*)d_in[12];
    const float* We       = (const float*)d_in[13];
    const float* att_be   = (const float*)d_in[14];
    const float* Wd       = (const float*)d_in[15];
    const float* att_bd   = (const float*)d_in[16];
    const float* Wv       = (const float*)d_in[17];
    const float* bv       = (const float*)d_in[18];
    const float* Wout     = (const float*)d_in[19];
    const float* bout     = (const float*)d_in[20];
    float*       dout     = (float*)d_out;

    // bump allocator over workspace (256B aligned slices)
    char* p = (char*)d_ws;
    auto alloc = [&](size_t bytes) -> char* {
        char* r = p;
        p += (bytes + 255) & ~(size_t)255;
        return r;
    };
    u16*   wcomb_e = (u16*)alloc((size_t)4 * H_ * 2 * H_ * 2);   // [4H, 2H] bf16
    u16*   wihd_b  = (u16*)alloc((size_t)4 * H_ * 2 * H_ * 2);   // [4H, 2H] bf16
    u16*   whhd_b  = (u16*)alloc((size_t)4 * H_ * H_ * 2);       // [4H, H]  bf16
    u16*   we_b    = (u16*)alloc((size_t)A_ * H_ * 2);
    u16*   wd_b    = (u16*)alloc((size_t)A_ * H_ * 2);
    u16*   wout_b  = (u16*)alloc((size_t)V_ * H_ * 2);           // 65.5 MB, L2-resident
    u16*   encb    = (u16*)alloc((size_t)B_ * S_ * H_ * 2);      // encoder_out bf16
    float* s1      = (float*)alloc((size_t)B_ * S_ * A_ * 4);    // hoisted enc projection
    float* gates   = (float*)alloc((size_t)B_ * 4 * H_ * 4);
    float* cbuf    = (float*)alloc((size_t)B_ * H_ * 4);         // c (-> cN after encoder)
    u16*   inpe    = (u16*)alloc((size_t)B_ * 2 * H_ * 2);       // [x_t ; h] bf16
    float* decrec  = (float*)alloc((size_t)B_ * 4 * H_ * 4);     // hN@Whh_d^T + b_d
    u16*   inpd    = (u16*)alloc((size_t)B_ * 2 * H_ * 2);       // [emb ; ctx] bf16
    u16*   hdb     = (u16*)alloc((size_t)B_ * H_ * 2);
    float* s2      = (float*)alloc((size_t)B_ * A_ * 4);
    float* be_     = (float*)alloc((size_t)4 * H_ * 4);
    float* bd_     = (float*)alloc((size_t)4 * H_ * 4);
    int*   tok     = (int*)alloc(256);

    // --- weight conversion to bf16 (once per launch, deterministic) ---
    build_wcomb<<<4096, 256, 0, stream>>>(Wih_e, Whh_e, wcomb_e);
    cvt_bf16<<<4096, 256, 0, stream>>>(Wih_d, wihd_b, (long)4 * H_ * 2 * H_);
    cvt_bf16<<<4096, 256, 0, stream>>>(Whh_d, whhd_b, (long)4 * H_ * H_);
    cvt_bf16<<<1024, 256, 0, stream>>>(We, we_b, (long)A_ * H_);
    cvt_bf16<<<1024, 256, 0, stream>>>(Wd, wd_b, (long)A_ * H_);
    cvt_bf16<<<4096, 256, 0, stream>>>(Wout, wout_b, (long)V_ * H_);
    addv<<<16, 256, 0, stream>>>(bih_e, bhh_e, be_, 4 * H_);
    addv<<<16, 256, 0, stream>>>(bih_d, bhh_d, bd_, 4 * H_);

    // --- encoder: fused gates = [x_t ; h] @ [Wih_e ; Whh_e]^T + b_e ---
    init_enc<<<B_, 256, 0, stream>>>(cbuf, inpe);
    for (int t = 0; t < S_; ++t) {
        enc_prep<<<B_, 256, 0, stream>>>(tok_enc, emb_enc, inpe, t);
        gemm_bf16<<<dim3(64, 1), 256, 0, stream>>>(
            inpe, 2 * H_, wcomb_e, 2 * H_, be_, nullptr, 0,
            gates, 4 * H_, B_, 4 * H_, 2 * H_);
        lstm_cell_enc<<<B_, 256, 0, stream>>>(gates, cbuf, inpe, encb, t);
    }

    // --- hoisted attention encoder projection: s1 = enc @ We^T + be ---
    gemm_bf16<<<dim3(A_ / 64, (B_ * S_) / 64), 256, 0, stream>>>(
        encb, H_, we_b, H_, att_be, nullptr, 0, s1, A_, B_ * S_, A_, H_);

    // --- hoisted decoder recurrent term: dec_rec = hN @ Whh_d^T + b_d ---
    gemm_bf16<<<dim3(64, 1), 256, 0, stream>>>(
        inpe + H_, 2 * H_, whhd_b, H_, bd_, nullptr, 0,
        decrec, 4 * H_, B_, 4 * H_, H_);

    // --- autoregressive decoder ---
    dec_init<<<B_, 256, 0, stream>>>(init_tok, init_ctx, tok, inpd);
    const long ldout = (long)ML_ * V_;
    for (int t = 0; t < ML_; ++t) {
        dec_prep<<<B_, 256, 0, stream>>>(tok, emb_dec, inpd);
        gemm_bf16<<<dim3(64, 1), 256, 0, stream>>>(
            inpd, 2 * H_, wihd_b, 2 * H_, nullptr, decrec, 4 * H_,
            gates, 4 * H_, B_, 4 * H_, 2 * H_);
        dec_cell<<<B_, 256, 0, stream>>>(gates, cbuf, hdb);
        gemm_bf16<<<dim3(A_ / 64, 1), 256, 0, stream>>>(
            hdb, H_, wd_b, H_, att_bd, nullptr, 0, s2, A_, B_, A_, H_);
        attention_kernel<<<B_, 256, 0, stream>>>(s1, s2, Wv, bv, encb, inpd);
        // logits straight into d_out[b, t, :]
        gemm_bf16<<<dim3((V_ + 63) / 64, 1), 256, 0, stream>>>(
            hdb, H_, wout_b, H_, bout, nullptr, 0,
            dout + (long)t * V_, (int)ldout, B_, V_, H_);
        argmax_kernel<<<B_, 256, 0, stream>>>(dout + (long)t * V_, ldout, tok, V_);
    }
}
